// CausalSelfAttention_4183298147119
// MI455X (gfx1250) — matrix operand order repaired and checked by reading, not compiled, not run on hardware
//
#include <hip/hip_runtime.h>
#include <hip/hip_bf16.h>
#include <math.h>

typedef __bf16 bf16;
typedef __attribute__((ext_vector_type(16))) __bf16 v16bf;
typedef __attribute__((ext_vector_type(8)))  __bf16 v8bf;
typedef __attribute__((ext_vector_type(8)))  float  v8f;

#define T_SEQ 2048
#define DIM   1024
#define NH    8
#define HD    128
#define ATTN_SCALE_C 0.12f

__device__ __forceinline__ unsigned short f2bf_bits(float f) {
  unsigned u = __builtin_bit_cast(unsigned, f);
  unsigned r = u + 0x7fffu + ((u >> 16) & 1u);   // round-to-nearest-even
  return (unsigned short)(r >> 16);
}
__device__ __forceinline__ bf16 f2bf(float f) {
  unsigned short s = f2bf_bits(f);
  return __builtin_bit_cast(bf16, s);
}
__device__ __forceinline__ unsigned pack2(unsigned short lo, unsigned short hi) {
  return (unsigned)lo | ((unsigned)hi << 16);
}

// ---------------------------------------------------------------- convert x
__global__ void cvt_f32_bf16_kernel(const float* __restrict__ in,
                                    bf16* __restrict__ out, int n) {
  int i = (blockIdx.x * blockDim.x + threadIdx.x) * 4;
  if (i + 3 < n) {
    float4 v = *(const float4*)(in + i);
    unsigned* o = (unsigned*)(out + i);
    o[0] = pack2(f2bf_bits(v.x), f2bf_bits(v.y));
    o[1] = pack2(f2bf_bits(v.z), f2bf_bits(v.w));
  }
}

// ---------------------------------------------------------------- GEMM
// C[2048,N] (f32) = A[2048,K] (bf16, row-major) @ B[N,K]^T (f32 weights).
// Block: 256 threads (8 waves). Tile: 128(M) x 128(N), K-step 32.
// Wave w owns 32x64: wm=(w>>1)*32, wn=(w&1)*64 -> 2x4 WMMA tiles (8 wmma/step).
// A tile staged via global_load_async_to_lds_b128 (ASYNCcnt); B converted
// f32->bf16 through VGPRs.
template <int N, int K>
__global__ __launch_bounds__(256) void gemm_bf16_wT_kernel(
    const bf16* __restrict__ A, const float* __restrict__ B,
    float* __restrict__ C) {
  __shared__ __attribute__((aligned(64))) bf16 As[128 * 32]; // [m][k]
  __shared__ __attribute__((aligned(64))) bf16 Bs[128 * 32]; // [n][k]
  const int tid  = threadIdx.x;
  const int lane = tid & 31;
  const int wave = tid >> 5;
  const int lidx = lane & 15;
  const int kgrp = lane >> 4;
  const int m0 = blockIdx.y * 128;
  const int n0 = blockIdx.x * 128;
  const int wm = (wave >> 1) * 32;
  const int wn = (wave & 1) * 64;

  v8f acc[2][4];
#pragma unroll
  for (int i = 0; i < 2; ++i)
#pragma unroll
    for (int j = 0; j < 4; ++j)
#pragma unroll
      for (int e = 0; e < 8; ++e) acc[i][j][e] = 0.0f;

  // staging assignments: thread -> (row, 16-element segment)
  const int srow = tid >> 1;
  const int sseg = (tid & 1) * 16;
  const bf16*  aptr = A + (size_t)(m0 + srow) * K + sseg;
  const float* bptr = B + (size_t)(n0 + srow) * K + sseg;
  const unsigned alds = (unsigned)(size_t)(As + srow * 32 + sseg); // LDS byte offset
  unsigned* blds = (unsigned*)(Bs + srow * 32 + sseg);

  for (int k0 = 0; k0 < K; k0 += 32) {
    __syncthreads();
    // A: direct global->LDS async copy (32 bytes per thread, bf16 source)
    asm volatile(
        "global_load_async_to_lds_b128 %0, %1, off\n\t"
        "global_load_async_to_lds_b128 %0, %1, off offset:16"
        :: "v"(alds + (unsigned)(k0 * 2)), "v"(aptr + k0)
        : "memory");
    // B: 16 fp32 -> 16 bf16, packed stores
    {
      const float* s = bptr + k0;
      float4 b0 = *(const float4*)(s + 0);
      float4 b1 = *(const float4*)(s + 4);
      float4 b2 = *(const float4*)(s + 8);
      float4 b3 = *(const float4*)(s + 12);
      uint4 w0, w1;
      w0.x = pack2(f2bf_bits(b0.x), f2bf_bits(b0.y));
      w0.y = pack2(f2bf_bits(b0.z), f2bf_bits(b0.w));
      w0.z = pack2(f2bf_bits(b1.x), f2bf_bits(b1.y));
      w0.w = pack2(f2bf_bits(b1.z), f2bf_bits(b1.w));
      w1.x = pack2(f2bf_bits(b2.x), f2bf_bits(b2.y));
      w1.y = pack2(f2bf_bits(b2.z), f2bf_bits(b2.w));
      w1.z = pack2(f2bf_bits(b3.x), f2bf_bits(b3.y));
      w1.w = pack2(f2bf_bits(b3.z), f2bf_bits(b3.w));
      *(uint4*)(blds + 0) = w0;
      *(uint4*)(blds + 4) = w1;
    }
    asm volatile("s_wait_asynccnt 0x0" ::: "memory");
    __syncthreads();

    v16bf af[2], bf[4];
#pragma unroll
    for (int i = 0; i < 2; ++i) { // A 16x32: lane=M row, 16 contiguous K at kgrp*16
      const bf16* ar = As + (wm + i * 16 + lidx) * 32;
      v8bf lo = *(const v8bf*)(ar + kgrp * 16);
      v8bf hi = *(const v8bf*)(ar + kgrp * 16 + 8);
#pragma unroll
      for (int e = 0; e < 8; ++e) { af[i][e] = lo[e]; af[i][8 + e] = hi[e]; }
    }
#pragma unroll
    for (int j = 0; j < 4; ++j)   // B 32x16: lane=N col, 16 contiguous K
      bf[j] = *(const v16bf*)(Bs + (wn + j * 16 + lidx) * 32 + kgrp * 16);
#pragma unroll
    for (int i = 0; i < 2; ++i)
#pragma unroll
      for (int j = 0; j < 4; ++j)
        acc[i][j] = __builtin_amdgcn_wmma_f32_16x16x32_bf16(
            false, af[i], false, bf[j], (short)0, acc[i][j], false, false);
  }

  // epilogue: one base pointer, compile-time strides -> immediate offsets
  float* cb = C + (size_t)(m0 + wm + kgrp * 8) * N + n0 + wn + lidx;
#pragma unroll
  for (int i = 0; i < 2; ++i)
#pragma unroll
    for (int j = 0; j < 4; ++j)
#pragma unroll
      for (int r = 0; r < 8; ++r)
        cb[(size_t)(i * 16 + r) * N + j * 16] = acc[i][j][r];
}

// ---------------------------------------------------------------- prep
// One wave per (t, h): RMS-norm q,k; rotary; v = l0*v + l1*ve.
// Lane owns d in {lane, lane+32, lane+64, lane+96}; rotary pairs (d, d+64).
__global__ __launch_bounds__(256) void prep_qkv_kernel(
    const float* __restrict__ qkv, const float* __restrict__ ve,
    const float* __restrict__ lambdas,
    bf16* __restrict__ Qb, bf16* __restrict__ Kb, bf16* __restrict__ Vt) {
  const int gw   = (blockIdx.x * blockDim.x + threadIdx.x) >> 5;
  const int lane = threadIdx.x & 31;
  const int t = gw >> 3;
  const int h = gw & 7;
  const float l0 = lambdas[0], l1 = lambdas[1];
  const float* base = qkv + (size_t)t * (3 * DIM);
  const float* qr = base + h * HD;
  const float* kr = base + (NH + h) * HD;
  const float* vr = base + (2 * NH + h) * HD;
  const float* ver = ve + (size_t)t * DIM + h * HD;

  float q[4], k[4];
#pragma unroll
  for (int e = 0; e < 4; ++e) {
    q[e] = qr[lane + e * 32];
    k[e] = kr[lane + e * 32];
  }
  float sq = q[0]*q[0] + q[1]*q[1] + q[2]*q[2] + q[3]*q[3];
  float sk = k[0]*k[0] + k[1]*k[1] + k[2]*k[2] + k[3]*k[3];
#pragma unroll
  for (int m = 16; m >= 1; m >>= 1) {
    sq += __shfl_xor(sq, m, 32);
    sk += __shfl_xor(sk, m, 32);
  }
  const float eps = 1.1920929e-07f;
  float rq = rsqrtf(sq * (1.0f / HD) + eps);
  float rk = rsqrtf(sk * (1.0f / HD) + eps);
#pragma unroll
  for (int e = 0; e < 4; ++e) { q[e] *= rq; k[e] *= rk; }

  // rotary: ang[j] = 2^(-10*j/31) for j<32 else 0; pairs (j, j+64). Lane = j.
  float ang = exp2f(-10.0f * (float)lane / 31.0f);
  float th = (float)t * ang;
  float c = cosf(th), s = sinf(th);
  float q0 = q[0] * c + q[2] * s, q2 = -q[0] * s + q[2] * c;
  float k0 = k[0] * c + k[2] * s, k2 = -k[0] * s + k[2] * c;
  q[0] = q0; q[2] = q2; k[0] = k0; k[2] = k2; // j in [32,64): identity (ang=0)

  bf16* qo = Qb + ((size_t)t * NH + h) * HD;
  bf16* ko = Kb + ((size_t)t * NH + h) * HD;
#pragma unroll
  for (int e = 0; e < 4; ++e) {
    int d = lane + e * 32;
    qo[d] = f2bf(q[e]);
    ko[d] = f2bf(k[e]);
    float vm = l0 * vr[d] + l1 * ver[d];
    Vt[((size_t)h * HD + d) * T_SEQ + t] = f2bf(vm); // V transposed [h][d][t]
  }
}

// ---------------------------------------------------------------- attention
// Grid (T/128, H). 8 independent waves/block; wave owns 16 query rows.
// Online softmax over 32-key chunks; P staged through per-wave LDS.
__global__ __launch_bounds__(256) void attn_kernel(
    const bf16* __restrict__ Qb, const bf16* __restrict__ Kb,
    const bf16* __restrict__ Vt, bf16* __restrict__ Yb) {
  __shared__ __attribute__((aligned(64))) bf16 pstage[8][16 * 32];
  const int wave = threadIdx.x >> 5;
  const int lane = threadIdx.x & 31;
  const int lidx = lane & 15;
  const int kgrp = lane >> 4;
  const int h = blockIdx.y;
  const int qr0 = blockIdx.x * 128 + wave * 16;

  v16bf qfrag[4]; // Q rows as 4 A-fragments over head dim
  {
    const bf16* qrow = Qb + ((size_t)(qr0 + lidx) * NH + h) * HD;
#pragma unroll
    for (int cI = 0; cI < 4; ++cI) {
      v8bf lo = *(const v8bf*)(qrow + cI * 32 + kgrp * 16);
      v8bf hi = *(const v8bf*)(qrow + cI * 32 + kgrp * 16 + 8);
#pragma unroll
      for (int e = 0; e < 8; ++e) { qfrag[cI][e] = lo[e]; qfrag[cI][8 + e] = hi[e]; }
    }
  }

  v8f o[8];
#pragma unroll
  for (int i = 0; i < 8; ++i)
#pragma unroll
    for (int e = 0; e < 8; ++e) o[i][e] = 0.0f;
  float rowM[8], rowL[8];
#pragma unroll
  for (int r = 0; r < 8; ++r) { rowM[r] = -__builtin_inff(); rowL[r] = 0.0f; }

  const int qmax = qr0 + 15;
  for (int kb = 0; kb <= qmax; kb += 32) {
    v8f s0, s1;
#pragma unroll
    for (int e = 0; e < 8; ++e) { s0[e] = 0.0f; s1[e] = 0.0f; }
#pragma unroll
    for (int cI = 0; cI < 4; ++cI) {
      v16bf b0 = *(const v16bf*)(Kb + ((size_t)(kb + lidx) * NH + h) * HD + cI * 32 + kgrp * 16);
      s0 = __builtin_amdgcn_wmma_f32_16x16x32_bf16(false, qfrag[cI], false, b0,
                                                   (short)0, s0, false, false);
      v16bf b1 = *(const v16bf*)(Kb + ((size_t)(kb + 16 + lidx) * NH + h) * HD + cI * 32 + kgrp * 16);
      s1 = __builtin_amdgcn_wmma_f32_16x16x32_bf16(false, qfrag[cI], false, b1,
                                                   (short)0, s1, false, false);
    }

    float a[8];
#pragma unroll
    for (int r = 0; r < 8; ++r) {
      int q = qr0 + r + kgrp * 8;
      s0[r] = ((kb + lidx)      <= q) ? s0[r] * ATTN_SCALE_C : -__builtin_inff();
      s1[r] = ((kb + 16 + lidx) <= q) ? s1[r] * ATTN_SCALE_C : -__builtin_inff();
      float mx = fmaxf(s0[r], s1[r]);
      mx = fmaxf(mx, __shfl_xor(mx, 1, 16));
      mx = fmaxf(mx, __shfl_xor(mx, 2, 16));
      mx = fmaxf(mx, __shfl_xor(mx, 4, 16));
      mx = fmaxf(mx, __shfl_xor(mx, 8, 16));
      float mnew = fmaxf(rowM[r], mx);
      a[r] = __expf(rowM[r] - mnew);
      rowM[r] = mnew;
      float p0 = __expf(s0[r] - mnew);
      float p1 = __expf(s1[r] - mnew);
      s0[r] = p0; s1[r] = p1;
      float rs = p0 + p1;
      rs += __shfl_xor(rs, 1, 16);
      rs += __shfl_xor(rs, 2, 16);
      rs += __shfl_xor(rs, 4, 16);
      rs += __shfl_xor(rs, 8, 16);
      rowL[r] = rowL[r] * a[r] + rs;
    }
#pragma unroll
    for (int dt = 0; dt < 8; ++dt)
#pragma unroll
      for (int r = 0; r < 8; ++r) o[dt][r] *= a[r];

    // stage P (16x32 bf16) and reload as an A-fragment
    bf16* ps = &pstage[wave][0];
#pragma unroll
    for (int r = 0; r < 8; ++r) {
      int m = r + kgrp * 8;
      ps[m * 32 + lidx]      = f2bf(s0[r]);
      ps[m * 32 + 16 + lidx] = f2bf(s1[r]);
    }
    asm volatile("s_wait_dscnt 0x0" ::: "memory");
    v16bf pf;
    {
      const bf16* pr = ps + lidx * 32;
      v8bf lo = *(const v8bf*)(pr + kgrp * 16);
      v8bf hi = *(const v8bf*)(pr + kgrp * 16 + 8);
#pragma unroll
      for (int e = 0; e < 8; ++e) { pf[e] = lo[e]; pf[8 + e] = hi[e]; }
    }
    // O += P @ V_chunk ; Vt is [h][d][t] so B-frags are contiguous per lane
#pragma unroll
    for (int dt = 0; dt < 8; ++dt) {
      v16bf vf = *(const v16bf*)(Vt + ((size_t)h * HD + dt * 16 + lidx) * T_SEQ + kb + kgrp * 16);
      o[dt] = __builtin_amdgcn_wmma_f32_16x16x32_bf16(false, pf, false, vf,
                                                      (short)0, o[dt], false, false);
    }
  }

#pragma unroll
  for (int r = 0; r < 8; ++r) rowL[r] = 1.0f / rowL[r];
#pragma unroll
  for (int dt = 0; dt < 8; ++dt)
#pragma unroll
    for (int r = 0; r < 8; ++r) {
      int m = r + kgrp * 8;
      Yb[(size_t)(qr0 + m) * DIM + h * HD + dt * 16 + lidx] = f2bf(o[dt][r] * rowL[r]);
    }
}

// ---------------------------------------------------------------- launch
extern "C" void kernel_launch(void* const* d_in, const int* in_sizes, int n_in,
                              void* d_out, int out_size, void* d_ws, size_t ws_size,
                              hipStream_t stream) {
  const float* x        = (const float*)d_in[0];
  const float* ve       = (const float*)d_in[1];
  const float* qkv_w    = (const float*)d_in[2]; // (3*1024, 1024) row-major
  const float* lambdas  = (const float*)d_in[3];
  const float* c_proj_w = (const float*)d_in[4]; // (1024, 1024) row-major
  float* out = (float*)d_out;

  char* ws = (char*)d_ws;
  bf16*  xb  = (bf16*)(ws);                       //  4 MB : x in bf16
  float* qkv = (float*)(ws + (4ull  << 20));      // 24 MB : qkv fp32 [t][3072]
  bf16*  Qb  = (bf16*)(ws + (28ull << 20));       //  4 MB : [t][h][d]
  bf16*  Kb  = (bf16*)(ws + (32ull << 20));       //  4 MB : [t][h][d]
  bf16*  Vt  = (bf16*)(ws + (36ull << 20));       //  4 MB : [h][d][t]
  bf16*  Yb  = (bf16*)(ws + (40ull << 20));       //  4 MB : [t][1024]

  const int nX = T_SEQ * DIM;
  cvt_f32_bf16_kernel<<<nX / 4 / 256, 256, 0, stream>>>(x, xb, nX);

  gemm_bf16_wT_kernel<3 * DIM, DIM><<<dim3(3 * DIM / 128, T_SEQ / 128), 256, 0, stream>>>(
      xb, qkv_w, qkv);

  prep_qkv_kernel<<<(T_SEQ * NH * 32) / 256, 256, 0, stream>>>(
      qkv, ve, lambdas, Qb, Kb, Vt);

  attn_kernel<<<dim3(T_SEQ / 128, NH), 256, 0, stream>>>(Qb, Kb, Vt, Yb);

  gemm_bf16_wT_kernel<DIM, DIM><<<dim3(DIM / 128, T_SEQ / 128), 256, 0, stream>>>(
      Yb, c_proj_w, out);
}
